// FederatedCausalNet_75746043232810
// MI455X (gfx1250) — matrix operand order repaired and checked by reading, not compiled, not run on hardware
//
#include <hip/hip_runtime.h>

// ---------------------------------------------------------------------------
// FederatedCausalNet for MI455X (gfx1250): mixed-precision WMMA pipeline.
//   - All O(n^3) block math on v_wmma_f32_16x16x32_f16 (f16 in, f32 acc)
//   - Global->LDS staging via GLOBAL_LOAD_ASYNC_TO_LDS_B128 (ASYNCcnt),
//     double-buffered; register-staged software pipeline as fallback
//   - fp32 VALU only for O(n*nb^2) panel factorization / per-row TRSM
//   - Transposed-RHS formulation => every big GEMM is NT (row-major x row-major^T)
// ---------------------------------------------------------------------------

#define N_PTS 4096
#define DX    64
#define DH    256
#define NB    64
#define NBLK  (N_PTS / NB)   // 64
#define MR    (N_PTS + 16)   // transposed RHS rows: row0 = y, rows16.. = Kmo

typedef __attribute__((ext_vector_type(16))) _Float16 v16h;
typedef __attribute__((ext_vector_type(8)))  _Float16 v8h;
typedef __attribute__((ext_vector_type(8)))  float    v8f;

#define F_SUBC 1   // C = C - acc   (beta=1 subtract epilogue)
#define F_BIAS 2   // acc += bias[col] (col < biasN)
#define F_RELU 4
#define F_HALF 8   // also mirror C to f16 buffer (next layer's A operand)

#if defined(__has_builtin)
#if __has_builtin(__builtin_amdgcn_global_load_async_to_lds_b128) && \
    __has_builtin(__builtin_amdgcn_s_wait_asynccnt)
#define HAVE_ASYNC_LDS 1
#endif
#endif
#ifndef HAVE_ASYNC_LDS
#define HAVE_ASYNC_LDS 0
#endif

#if HAVE_ASYNC_LDS
// builtin signature (per hipcc diagnostic): (v4i AS1* src, v4i AS3* dst, Ii offset, Ii cpol)
typedef int v4i_g __attribute__((vector_size(16)));
#define GPTR(p) ((__attribute__((address_space(1))) v4i_g*)(p))
#define LPTR(p) ((__attribute__((address_space(3))) v4i_g*)(p))
#endif

// ---------------------------------------------------------------------------
// NT GEMM: C[M,N] (+)= A[M,K] * B[N,K]^T, f16 operands, f32 accumulate.
// 256 threads = 8 waves; TM=128 x TN=(128|64) C tile; K-step 32;
// double-buffered LDS, async global->LDS staging.
// ---------------------------------------------------------------------------
template <int TN>
__global__ __launch_bounds__(256) void gemm_nt_wmma(
    const _Float16* __restrict__ A, int lda,
    const _Float16* __restrict__ B, int ldb,
    float* __restrict__ C, int ldc,
    int M, int N, int Kd, int flags,
    const float* __restrict__ bias, int biasN,
    _Float16* __restrict__ Ch, int ldch)
{
  constexpr int TM = 128;
  constexpr int WN = (TN == 128) ? 2 : 1;  // waves along N
  constexpr int MI = (TN == 128) ? 2 : 1;  // 16-row tiles per wave
  constexpr int NI = 4;                    // 16-col tiles per wave (64 cols)
  constexpr int BOPS = (TN == 128) ? 2 : 1;
  constexpr int OPS  = 2 + BOPS;           // async b128 ops per thread per tile

  __shared__ __align__(64) _Float16 sA[2][TM][32];
  __shared__ __align__(64) _Float16 sB[2][TN][32];

  const int tid  = threadIdx.x;
  const int wave = tid >> 5;
  const int lane = tid & 31;
  const int bm = blockIdx.y * TM;
  const int bn = blockIdx.x * TN;
  const int wm = wave / WN;
  const int wn = wave % WN;

  v8f acc[MI][NI];
#pragma unroll
  for (int mi = 0; mi < MI; ++mi)
#pragma unroll
    for (int nt = 0; nt < NI; ++nt) acc[mi][nt] = (v8f){};

  // staging: A tile 128x32 halves -> 32B/thread; B tile TNx32 halves
  const int sra = tid >> 1;
  const int soa = (tid & 1) * 16;
  int gar = bm + sra; if (gar > M - 1) gar = M - 1;
  const _Float16* pa = A + (size_t)gar * lda + soa;

  const int srb = (TN == 128) ? (tid >> 1) : (tid >> 2);
  const int sob = (TN == 128) ? ((tid & 1) * 16) : ((tid & 3) * 8);
  int gbr = bn + srb; if (gbr > N - 1) gbr = N - 1;
  const _Float16* pb = B + (size_t)gbr * ldb + sob;

  // WMMA fragment addressing (ISA 16-bit A 16x32 / B 32x16 layouts)
  const int ar = wm * (MI * 16) + (lane & 15);
  const int ak = (lane >> 4) << 3;   // 0 or 8
  const int bc = wn * 64 + (lane & 15);
  const int bk = (lane >> 4) << 3;   // 0 or 8

  auto compute = [&](int buf) {
    v16h af[MI];
#pragma unroll
    for (int mi = 0; mi < MI; ++mi) {
      v8h lo = *(const v8h*)&sA[buf][ar + mi * 16][ak];
      v8h hi = *(const v8h*)&sA[buf][ar + mi * 16][ak + 16];
      af[mi] = __builtin_shufflevector(lo, hi, 0,1,2,3,4,5,6,7,8,9,10,11,12,13,14,15);
    }
#pragma unroll
    for (int nt = 0; nt < NI; ++nt) {
      v8h blo = *(const v8h*)&sB[buf][bc + nt * 16][bk];
      v8h bhi = *(const v8h*)&sB[buf][bc + nt * 16][bk + 16];
      v16h bf = __builtin_shufflevector(blo, bhi, 0,1,2,3,4,5,6,7,8,9,10,11,12,13,14,15);
#pragma unroll
      for (int mi = 0; mi < MI; ++mi)
        acc[mi][nt] = __builtin_amdgcn_wmma_f32_16x16x32_f16(
            false, af[mi], false, bf, (short)0, acc[mi][nt], false, false);
    }
  };

#if HAVE_ASYNC_LDS
  // ---- async global->LDS double-buffered pipeline (ASYNCcnt) ----
  auto issue = [&](int buf, int k0) {
    __builtin_amdgcn_global_load_async_to_lds_b128(GPTR(pa + k0),     LPTR(&sA[buf][sra][soa]),     0, 0);
    __builtin_amdgcn_global_load_async_to_lds_b128(GPTR(pa + k0 + 8), LPTR(&sA[buf][sra][soa + 8]), 0, 0);
    __builtin_amdgcn_global_load_async_to_lds_b128(GPTR(pb + k0),     LPTR(&sB[buf][srb][sob]),     0, 0);
    if (BOPS == 2)
      __builtin_amdgcn_global_load_async_to_lds_b128(GPTR(pb + k0 + 8), LPTR(&sB[buf][srb][sob + 8]), 0, 0);
  };
  int cur = 0;
  int k0 = 0;
  issue(0, 0);
  for (; k0 + 32 < Kd; k0 += 32) {
    issue(cur ^ 1, k0 + 32);                    // prefetch next tile into other buffer
    __builtin_amdgcn_s_wait_asynccnt(OPS);      // current tile's transfers done (in-order)
    __syncthreads();
    compute(cur);
    __syncthreads();                            // all waves done reading before overwrite
    cur ^= 1;
  }
  __builtin_amdgcn_s_wait_asynccnt(0);
  __syncthreads();
  compute(cur);
#else
  // ---- fallback: register-staged software pipeline ----
  uint4 ra0, ra1, rb0, rb1;
  auto gload = [&](int k0) {
    ra0 = *(const uint4*)(pa + k0);
    ra1 = *(const uint4*)(pa + k0 + 8);
    rb0 = *(const uint4*)(pb + k0);
    if (BOPS == 2) rb1 = *(const uint4*)(pb + k0 + 8);
    if (k0 + 32 < Kd) {
      __builtin_prefetch(pa + k0 + 32, 0, 1);   // global_prefetch_b8
      __builtin_prefetch(pb + k0 + 32, 0, 1);
    }
  };
  auto sstore = [&](int buf) {
    *(uint4*)&sA[buf][sra][soa]     = ra0;
    *(uint4*)&sA[buf][sra][soa + 8] = ra1;
    *(uint4*)&sB[buf][srb][sob]     = rb0;
    if (BOPS == 2) *(uint4*)&sB[buf][srb][sob + 8] = rb1;
  };
  gload(0);
  sstore(0);
  int cur = 0;
  int k0 = 0;
  for (; k0 + 32 < Kd; k0 += 32) {
    __syncthreads();          // sA/sB[cur] visible to all waves
    gload(k0 + 32);           // issue next tile's global loads (consumed after compute)
    compute(cur);
    sstore(cur ^ 1);          // write other buffer; readers gated by next barrier
    cur ^= 1;
  }
  __syncthreads();
  compute(cur);
#endif

  // C/D layout: lanes 0-15 -> M=v, lanes 16-31 -> M=8+v; N = lane&15
  const int rbase = bm + wm * (MI * 16) + ((lane >> 4) << 3);
  const int cbase = bn + wn * 64 + (lane & 15);
#pragma unroll
  for (int mi = 0; mi < MI; ++mi) {
#pragma unroll
    for (int nt = 0; nt < NI; ++nt) {
      int col = cbase + nt * 16;
      if (col >= N) continue;
#pragma unroll
      for (int v = 0; v < 8; ++v) {
        int row = rbase + mi * 16 + v;
        if (row >= M) continue;
        float val = acc[mi][nt][v];
        size_t ci = (size_t)row * ldc + col;
        if (flags & F_SUBC) val = C[ci] - val;
        if (flags & F_BIAS) val += (col < biasN) ? bias[col] : 0.0f;
        if (flags & F_RELU) val = fmaxf(val, 0.0f);
        C[ci] = val;
        if (flags & F_HALF) Ch[(size_t)row * ldch + col] = (_Float16)val;
      }
    }
  }
}

// ---------------------------------------------------------------------------
// Small / elementwise kernels (fp32)
// ---------------------------------------------------------------------------
__global__ void prep_small(const float* __restrict__ ph, const float* __restrict__ sh,
                           float* __restrict__ scal)
{
  if (threadIdx.x == 0 && blockIdx.x == 0) {
    float L00 = ph[0], L10 = ph[2], L11 = ph[3];   // tril(phi_half)
    float S00 = sh[0], S10 = sh[2], S11 = sh[3];   // tril(sig_half)
    scal[0] = L00 * L00;                // Phi00
    scal[1] = L00 * L10;                // Phi01 = Phi10
    scal[2] = L10 * L10 + L11 * L11;    // Phi11
    scal[3] = S00 * S00;                // Sig00
    scal[4] = S00 * S10;                // Sig01 = Sig10
    scal[5] = S10 * S10 + S11 * S11;    // Sig11
    scal[6] = L00; scal[7] = L10; scal[8] = L11;
  }
}

__global__ void convert_x(const float* __restrict__ X, const float* __restrict__ ls,
                          _Float16* __restrict__ x16, _Float16* __restrict__ a16, int total)
{
  int i = blockIdx.x * blockDim.x + threadIdx.x;
  if (i >= total) return;
  float inv = 1.0f / ls[0];
  float v = X[i];
  x16[i] = (_Float16)v;
  a16[i] = (_Float16)(v * inv);
}

__global__ void sqnorm_rows(const float* __restrict__ X, const float* __restrict__ ls,
                            float* __restrict__ sq)
{
  int r = blockIdx.x * blockDim.x + threadIdx.x;
  if (r >= N_PTS) return;
  float inv = 1.0f / ls[0];
  const float* row = X + (size_t)r * DX;
  float s = 0.f;
  for (int k = 0; k < DX; ++k) { float a = row[k] * inv; s += a * a; }
  sq[r] = s;
}

// W (din x dout, f32, row-major) -> W^T (dpad x din, f16), zero-padded rows
__global__ void wtrans(const float* __restrict__ W, int din, int dout, int dpad,
                       _Float16* __restrict__ dst)
{
  int idx = blockIdx.x * blockDim.x + threadIdx.x;
  if (idx >= dpad * din) return;
  int o = idx / din, i = idx % din;
  dst[idx] = (o < dout) ? (_Float16)W[(size_t)i * dout + o] : (_Float16)0.0f;
}

// gram -> Matern 2.5 kernel; assemble Kobs / Kmis(->d_out) / Kmo / Kmo16 / rhs^T rows
__global__ void matern_assemble(const float* __restrict__ gram, const float* __restrict__ sq,
                                const int* __restrict__ Wint, const float* __restrict__ scal,
                                float* __restrict__ Kobs, float* __restrict__ Kmo,
                                _Float16* __restrict__ Kmo16, float* __restrict__ R,
                                float* __restrict__ Smis, int n)
{
  size_t idx = (size_t)blockIdx.x * blockDim.x + threadIdx.x;
  if (idx >= (size_t)n * n) return;
  int i = (int)(idx / n), j = (int)(idx % n);
  float p00 = scal[0], p01 = scal[1], p11 = scal[2];
  float s00 = scal[3], s01 = scal[4], s11 = scal[5];

  float g  = gram[idx];
  float d2 = fmaxf(sq[i] + sq[j] - 2.0f * g, 0.0f);
  float d  = sqrtf(d2 + 1e-12f);
  const float S5 = 2.2360679774997896f;
  float Kv = (1.0f + S5 * d + 1.6666666666666667f * d2) * __expf(-S5 * d);

  float ai = (float)Wint[i], aj = (float)Wint[j];
  float nai = 1.0f - ai, naj = 1.0f - aj;
  float t0o = p00 * naj + p01 * aj,  t1o = p01 * naj + p11 * aj;   // Phi * bo_j
  float t0m = p00 * aj  + p01 * naj, t1m = p01 * aj  + p11 * naj;  // Phi * bm_j
  float kobs = (nai * t0o + ai * t1o) * Kv;
  float kmis = (ai * t0m + nai * t1m) * Kv;
  float kmo  = (ai * t0o + nai * t1o) * Kv;
  if (i == j) {
    kobs += nai * nai * s00 + ai * ai * s11;
    kmis += ai * ai * s00 + nai * nai * s11;
    kmo  += s01;                       // (na+a)*Sig01 = Sig01
  }
  Kobs[idx]  = kobs;
  Smis[idx]  = kmis;
  Kmo[idx]   = kmo;
  Kmo16[idx] = (_Float16)kmo;
  R[(size_t)(16 + i) * n + j] = kmo;   // rhs^T rows 16.. = Kmo
}

__global__ void mu_combine(const float* __restrict__ mu0m, const float* __restrict__ mu1m,
                           const int* __restrict__ Wint, const float* __restrict__ Yobs,
                           const float* __restrict__ scal, float* __restrict__ muMis,
                           float* __restrict__ R, int n)
{
  int i = blockIdx.x * blockDim.x + threadIdx.x;
  if (i >= n) return;
  float L00 = scal[6], L10 = scal[7], L11 = scal[8];
  float w  = (float)Wint[i];
  float m0 = mu0m[(size_t)i * 16], m1 = mu1m[(size_t)i * 16];
  float c  = L10 * m0 + L11 * m1;
  float muO = (1.0f - w) * L00 * m0 + w * c;
  muMis[i]  = w * L00 * m0 + (1.0f - w) * c;
  R[i] = Yobs[i] - muO;                                  // row 0 = y^T
  for (int p = 1; p < 16; ++p) R[(size_t)p * n + i] = 0.0f;  // padding rows
}

// in-place 64x64 fp32 Cholesky on the diagonal block (single 64-thread block)
__global__ __launch_bounds__(64) void potf64(float* __restrict__ Ablk, int ld)
{
  __shared__ float s[64][65];
  int t = threadIdx.x;
  for (int u = 0; u < 64; ++u) s[t][u] = Ablk[(size_t)t * ld + u];
  __syncthreads();
  for (int c = 0; c < 64; ++c) {
    if (t == c) s[c][c] = sqrtf(s[c][c]);
    __syncthreads();
    if (t > c) s[t][c] /= s[c][c];
    __syncthreads();
    if (t > c) {
      float f = s[t][c];
      for (int u = c + 1; u <= t; ++u) s[t][u] -= f * s[u][c];
    }
    __syncthreads();
  }
  for (int u = 0; u < 64; ++u) Ablk[(size_t)t * ld + u] = (u <= t) ? s[t][u] : 0.0f;
}

// per-row solves of x * Ljj^T = r  (forward over 64 cols); Rb pre-offset to col block
__global__ __launch_bounds__(64) void row_trsm_fwd(float* __restrict__ Rb, int ldr, int nrows,
                                                   const float* __restrict__ Ld, int ldl)
{
  __shared__ float xs[64][NB + 1];
  int m = blockIdx.x * 64 + threadIdx.x;
  if (m >= nrows) return;
  float* r = Rb + (size_t)m * ldr;
  int t0 = threadIdx.x;
  for (int c = 0; c < NB; ++c) xs[t0][c] = r[c];
  for (int c = 0; c < NB; ++c) {
    float s = xs[t0][c];
    for (int t = 0; t < c; ++t) s -= xs[t0][t] * Ld[(size_t)c * ldl + t];
    xs[t0][c] = s / Ld[(size_t)c * ldl + c];
  }
  for (int c = 0; c < NB; ++c) r[c] = xs[t0][c];
}

// per-row solves of x * Ljj = r  (backward over 64 cols)
__global__ __launch_bounds__(64) void row_trsm_bwd(float* __restrict__ Rb, int ldr, int nrows,
                                                   const float* __restrict__ Ld, int ldl)
{
  __shared__ float xs[64][NB + 1];
  int m = blockIdx.x * 64 + threadIdx.x;
  if (m >= nrows) return;
  float* r = Rb + (size_t)m * ldr;
  int t0 = threadIdx.x;
  for (int c = 0; c < NB; ++c) xs[t0][c] = r[c];
  for (int c = NB - 1; c >= 0; --c) {
    float s = xs[t0][c];
    for (int t = c + 1; t < NB; ++t) s -= xs[t0][t] * Ld[(size_t)t * ldl + c];
    xs[t0][c] = s / Ld[(size_t)c * ldl + c];
  }
  for (int c = 0; c < NB; ++c) r[c] = xs[t0][c];
}

// f32 block -> f16 (optionally transposed: produces L^T panels for the backward sweep)
__global__ void f2h_block(const float* __restrict__ src, int lds_, _Float16* __restrict__ dst,
                          int ldd, int rows, int cols, int transp)
{
  size_t idx = (size_t)blockIdx.x * blockDim.x + threadIdx.x;
  if (idx >= (size_t)rows * cols) return;
  int r = (int)(idx / cols), c = (int)(idx % cols);
  float v = src[(size_t)r * lds_ + c];
  if (transp) dst[(size_t)c * ldd + r] = (_Float16)v;
  else        dst[(size_t)r * ldd + c] = (_Float16)v;
}

// m_mo = muMis + Kmo * y  (block per row, fp32)
__global__ __launch_bounds__(256) void matvec_m(const float* __restrict__ Kmo,
                                                const float* __restrict__ y,
                                                const float* __restrict__ muMis,
                                                float* __restrict__ out, int n)
{
  __shared__ float red[256];
  int row = blockIdx.x;
  const float* kr = Kmo + (size_t)row * n;
  float s = 0.f;
  for (int k = threadIdx.x; k < n; k += 256) s += kr[k] * y[k];
  red[threadIdx.x] = s;
  __syncthreads();
  for (int st = 128; st > 0; st >>= 1) {
    if (threadIdx.x < st) red[threadIdx.x] += red[threadIdx.x + st];
    __syncthreads();
  }
  if (threadIdx.x == 0) out[row] = muMis[row] + red[0];
}

// ---------------------------------------------------------------------------
// Host side
// ---------------------------------------------------------------------------
static void launch_gemm(hipStream_t st, const _Float16* A, int lda, const _Float16* B, int ldb,
                        float* C, int ldc, int M, int N, int K, int flags,
                        const float* bias, int biasN, _Float16* Ch, int ldch)
{
  dim3 b(256);
  if (N > 64) {
    dim3 g((N + 127) / 128, (M + 127) / 128);
    gemm_nt_wmma<128><<<g, b, 0, st>>>(A, lda, B, ldb, C, ldc, M, N, K, flags, bias, biasN, Ch, ldch);
  } else {
    dim3 g(1, (M + 127) / 128);
    gemm_nt_wmma<64><<<g, b, 0, st>>>(A, lda, B, ldb, C, ldc, M, N, K, flags, bias, biasN, Ch, ldch);
  }
}

extern "C" void kernel_launch(void* const* d_in, const int* in_sizes, int n_in,
                              void* d_out, int out_size, void* d_ws, size_t ws_size,
                              hipStream_t stream)
{
  (void)in_sizes; (void)n_in; (void)out_size;
  const float* X    = (const float*)d_in[0];
  const float* Yobs = (const float*)d_in[1];
  const int*   Wint = (const int*)  d_in[2];
  const float* ph   = (const float*)d_in[3];
  const float* sh   = (const float*)d_in[4];
  const float* ls   = (const float*)d_in[5];
  const float* ws0 = (const float*)d_in[6],  *bs0 = (const float*)d_in[7];
  const float* ws1 = (const float*)d_in[8],  *bs1 = (const float*)d_in[9];
  const float* ws2 = (const float*)d_in[10], *bs2 = (const float*)d_in[11];
  const float* w00 = (const float*)d_in[12], *b00 = (const float*)d_in[13];
  const float* w01 = (const float*)d_in[14], *b01 = (const float*)d_in[15];
  const float* w02 = (const float*)d_in[16], *b02 = (const float*)d_in[17];
  const float* w10 = (const float*)d_in[18], *b10 = (const float*)d_in[19];
  const float* w11 = (const float*)d_in[20], *b11 = (const float*)d_in[21];
  const float* w12 = (const float*)d_in[22], *b12 = (const float*)d_in[23];
  float* dout = (float*)d_out;

  const size_t nn = (size_t)N_PTS * N_PTS;
  char* w = (char*)d_ws;
  size_t off = 0;
  auto WS = [&](size_t bytes) { size_t o = off; off += (bytes + 255) & ~(size_t)255; return o; };

  float*    scal  = (float*)   (w + WS(64 * 4));
  _Float16* a16   = (_Float16*)(w + WS((size_t)N_PTS * DX * 2));
  _Float16* x16   = (_Float16*)(w + WS((size_t)N_PTS * DX * 2));
  float*    sq    = (float*)   (w + WS((size_t)N_PTS * 4));
  float*    gram  = (float*)   (w + WS(nn * 4));
  float*    Kobs  = (float*)   (w + WS(nn * 4));   // becomes L after factorization
  float*    Kmo   = (float*)   (w + WS(nn * 4));
  _Float16* Kmo16 = (_Float16*)(w + WS(nn * 2));
  float*    R     = (float*)   (w + WS((size_t)MR * N_PTS * 4));   // rhs^T, then Z^T, then sol^T
  _Float16* Rh    = (_Float16*)(w + WS((size_t)MR * N_PTS * 2));   // f16 mirror of solved blocks
  _Float16* Lh    = (_Float16*)(w + WS(nn * 2));   // f16 L
  _Float16* LhT   = (_Float16*)(w + WS(nn * 2));   // f16 L^T
  float*    hbuf  = (float*)   (w + WS((size_t)N_PTS * DH * 4));
  _Float16* h16a  = (_Float16*)(w + WS((size_t)N_PTS * DH * 2));
  _Float16* h16b  = (_Float16*)(w + WS((size_t)N_PTS * DH * 2));
  _Float16* t16   = (_Float16*)(w + WS((size_t)N_PTS * DH * 2));
  float*    mu0m  = (float*)   (w + WS((size_t)N_PTS * 16 * 4));
  float*    mu1m  = (float*)   (w + WS((size_t)N_PTS * 16 * 4));
  float*    muMis = (float*)   (w + WS((size_t)N_PTS * 4));
  _Float16* ws0T  = (_Float16*)(w + WS((size_t)DH * DX * 2));
  _Float16* ws1T  = (_Float16*)(w + WS((size_t)DH * DH * 2));
  _Float16* ws2T  = (_Float16*)(w + WS((size_t)DH * DH * 2));
  _Float16* w00T  = (_Float16*)(w + WS((size_t)DH * DH * 2));
  _Float16* w01T  = (_Float16*)(w + WS((size_t)DH * DH * 2));
  _Float16* w10T  = (_Float16*)(w + WS((size_t)DH * DH * 2));
  _Float16* w11T  = (_Float16*)(w + WS((size_t)DH * DH * 2));
  _Float16* w02T  = (_Float16*)(w + WS((size_t)16 * DH * 2));
  _Float16* w12T  = (_Float16*)(w + WS((size_t)16 * DH * 2));
  if (ws_size < off) return;   // workspace too small

  // --- prep ---------------------------------------------------------------
  prep_small<<<1, 32, 0, stream>>>(ph, sh, scal);
  {
    int tot = N_PTS * DX;
    convert_x<<<(tot + 255) / 256, 256, 0, stream>>>(X, ls, x16, a16, tot);
    sqnorm_rows<<<(N_PTS + 255) / 256, 256, 0, stream>>>(X, ls, sq);
  }
  auto wt = [&](const float* Ww, int din, int dout_, int dpad, _Float16* dst) {
    int tot = dpad * din;
    wtrans<<<(tot + 255) / 256, 256, 0, stream>>>(Ww, din, dout_, dpad, dst);
  };
  wt(ws0, DX, DH, DH, ws0T);  wt(ws1, DH, DH, DH, ws1T);  wt(ws2, DH, DH, DH, ws2T);
  wt(w00, DH, DH, DH, w00T);  wt(w01, DH, DH, DH, w01T);  wt(w02, DH, 1, 16, w02T);
  wt(w10, DH, DH, DH, w10T);  wt(w11, DH, DH, DH, w11T);  wt(w12, DH, 1, 16, w12T);

  // --- gram + Matern assembly ---------------------------------------------
  launch_gemm(stream, a16, DX, a16, DX, gram, N_PTS, N_PTS, N_PTS, DX, 0, nullptr, 0, nullptr, 0);
  {
    size_t tot = nn;
    matern_assemble<<<(unsigned)((tot + 255) / 256), 256, 0, stream>>>(
        gram, sq, Wint, scal, Kobs, Kmo, Kmo16, R, dout + N_PTS, N_PTS);
  }

  // --- MLPs (bias+ReLU fused into GEMM epilogue) ---------------------------
  launch_gemm(stream, x16,  DX, ws0T, DX, hbuf, DH, N_PTS, DH, DX, F_BIAS|F_RELU|F_HALF, bs0, DH, h16a, DH);
  launch_gemm(stream, h16a, DH, ws1T, DH, hbuf, DH, N_PTS, DH, DH, F_BIAS|F_RELU|F_HALF, bs1, DH, h16b, DH);
  launch_gemm(stream, h16b, DH, ws2T, DH, hbuf, DH, N_PTS, DH, DH, F_BIAS|F_RELU|F_HALF, bs2, DH, h16a, DH);
  // head 0
  launch_gemm(stream, h16a, DH, w00T, DH, hbuf, DH, N_PTS, DH, DH, F_BIAS|F_RELU|F_HALF, b00, DH, h16b, DH);
  launch_gemm(stream, h16b, DH, w01T, DH, hbuf, DH, N_PTS, DH, DH, F_BIAS|F_RELU|F_HALF, b01, DH, t16,  DH);
  launch_gemm(stream, t16,  DH, w02T, DH, mu0m, 16, N_PTS, 16, DH, F_BIAS, b02, 1, nullptr, 0);
  // head 1
  launch_gemm(stream, h16a, DH, w10T, DH, hbuf, DH, N_PTS, DH, DH, F_BIAS|F_RELU|F_HALF, b10, DH, h16b, DH);
  launch_gemm(stream, h16b, DH, w11T, DH, hbuf, DH, N_PTS, DH, DH, F_BIAS|F_RELU|F_HALF, b11, DH, t16,  DH);
  launch_gemm(stream, t16,  DH, w12T, DH, mu1m, 16, N_PTS, 16, DH, F_BIAS, b12, 1, nullptr, 0);

  mu_combine<<<(N_PTS + 255) / 256, 256, 0, stream>>>(mu0m, mu1m, Wint, Yobs, scal, muMis, R, N_PTS);

  // --- blocked left-looking Cholesky of Kobs (L overwrites lower triangle) --
  for (int j = 0; j < NBLK; ++j) {
    int r0 = j * NB;
    int mrem = N_PTS - r0;
    if (j > 0)
      launch_gemm(stream, Lh + (size_t)r0 * N_PTS, N_PTS, Lh + (size_t)r0 * N_PTS, N_PTS,
                  Kobs + (size_t)r0 * N_PTS + r0, N_PTS, mrem, NB, r0, F_SUBC, nullptr, 0, nullptr, 0);
    potf64<<<1, 64, 0, stream>>>(Kobs + (size_t)r0 * N_PTS + r0, N_PTS);
    if (mrem > NB)
      row_trsm_fwd<<<(mrem - NB + 63) / 64, 64, 0, stream>>>(
          Kobs + (size_t)(r0 + NB) * N_PTS + r0, N_PTS, mrem - NB,
          Kobs + (size_t)r0 * N_PTS + r0, N_PTS);
    int tot = mrem * NB;
    f2h_block<<<(tot + 255) / 256, 256, 0, stream>>>(
        Kobs + (size_t)r0 * N_PTS + r0, N_PTS, Lh + (size_t)r0 * N_PTS + r0, N_PTS, mrem, NB, 0);
    f2h_block<<<(tot + 255) / 256, 256, 0, stream>>>(
        Kobs + (size_t)r0 * N_PTS + r0, N_PTS, LhT + (size_t)r0 * N_PTS + r0, N_PTS, mrem, NB, 1);
  }

  // --- forward sweep: Z^T * L^T = rhs^T  (left-looking over column blocks) --
  for (int j = 0; j < NBLK; ++j) {
    int r0 = j * NB;
    if (j > 0)
      launch_gemm(stream, Rh, N_PTS, Lh + (size_t)r0 * N_PTS, N_PTS,
                  R + r0, N_PTS, MR, NB, r0, F_SUBC, nullptr, 0, nullptr, 0);
    row_trsm_fwd<<<(MR + 63) / 64, 64, 0, stream>>>(
        R + r0, N_PTS, MR, Kobs + (size_t)r0 * N_PTS + r0, N_PTS);
    int tot = MR * NB;
    f2h_block<<<(tot + 255) / 256, 256, 0, stream>>>(R + r0, N_PTS, Rh + r0, N_PTS, MR, NB, 0);
  }

  // --- backward sweep: sol^T * L = Z^T -------------------------------------
  for (int j = NBLK - 1; j >= 0; --j) {
    int r0 = j * NB;
    int kk = N_PTS - r0 - NB;
    if (kk > 0)
      launch_gemm(stream, Rh + (r0 + NB), N_PTS, LhT + (size_t)r0 * N_PTS + (r0 + NB), N_PTS,
                  R + r0, N_PTS, MR, NB, kk, F_SUBC, nullptr, 0, nullptr, 0);
    row_trsm_bwd<<<(MR + 63) / 64, 64, 0, stream>>>(
        R + r0, N_PTS, MR, Kobs + (size_t)r0 * N_PTS + r0, N_PTS);
    int tot = MR * NB;
    f2h_block<<<(tot + 255) / 256, 256, 0, stream>>>(R + r0, N_PTS, Rh + r0, N_PTS, MR, NB, 0);
  }

  // --- S_mo = Kmis - Kmo * sol[:,1:]  (NT: B = sol^T rows 16..) -------------
  launch_gemm(stream, Kmo16, N_PTS, Rh + (size_t)16 * N_PTS, N_PTS,
              dout + N_PTS, N_PTS, N_PTS, N_PTS, N_PTS, F_SUBC, nullptr, 0, nullptr, 0);

  // --- m_mo = muMis + Kmo * sol[:,0]  (sol^T row 0 lives in R row 0) --------
  matvec_m<<<N_PTS, 256, 0, stream>>>(Kmo, R, muMis, dout, N_PTS);
}
